// LSTMSeq2Seq_2336462209463
// MI455X (gfx1250) — compile-verified, matrix operand order checked by reading, not run on hardware
//
#include <hip/hip_runtime.h>
#include <hip/hip_bf16.h>

// ---------------------------------------------------------------------------
// LSTM seq2seq on MI455X (gfx1250), bf16 WMMA path.
//   B=64, T=128, D=512, H=1024, FUT=128 (prediction_len fixed; device scalar
//   cannot be read host-side under graph capture).
// Weights converted once per call to bf16 [N x K] (transposed): 63 MB total,
// resident in the 192 MB L2 across all 512 recurrent steps. Each LSTM layer
// step is one fused kernel: WMMA GEMM over [x|h] with 4 gate accumulators per
// wave + sigmoid/tanh/c/h epilogue. The K-loop is split per input segment,
// unrolled by two 32-wide chunks with two fragment sets defined directly by
// loads (no register rotation movs), and the tail chunk is peeled so the
// steady-state loop is branch-free with ~10 B128 loads in flight per 4-WMMA
// group.
// ---------------------------------------------------------------------------

typedef __attribute__((ext_vector_type(16))) __bf16 v16bf;
typedef __attribute__((ext_vector_type(8)))  float  v8f;

#define BATCH 64
#define SEQT  128
#define DIN   512
#define HID   1024
#define FUT   128

__device__ __forceinline__ unsigned short f32_to_bf16(float f) {
    unsigned u = __float_as_uint(f);
    unsigned rnd = 0x7FFFu + ((u >> 16) & 1u);   // round-to-nearest-even
    return (unsigned short)((u + rnd) >> 16);
}
__device__ __forceinline__ float sigmoidf_(float x) {
    return 1.0f / (1.0f + __expf(-x));
}
__device__ __forceinline__ float tanhf_(float x) {
    return 2.0f * sigmoidf_(2.0f * x) - 1.0f;
}

// Load a 16x32 bf16 WMMA fragment from a row-major tile.
// ISA layout (CDNA5 ISA): lanes 0-15 hold row r, K={0..7,16..23};
// lanes 16-31 hold row r, K={8..15,24..31}. Two B128 loads per lane.
__device__ __forceinline__ v16bf load_frag16x32(const unsigned short* __restrict__ base,
                                                int row_stride, int lane, int k0) {
    int r   = lane & 15;
    int sel = (lane >> 4) & 1;
    const unsigned short* p = base + (size_t)r * row_stride + k0 + sel * 8;
    union { v16bf v; uint4 q[2]; } u;
    u.q[0] = *(const uint4*)(p);
    u.q[1] = *(const uint4*)(p + 16);
    return u.v;
}

// One K-chunk's worth of operands: A fragment + 4 gate B fragments (40 VGPRs).
struct FragSet { v16bf a, b0, b1, b2, b3; };

__device__ __forceinline__ FragSet load_set(const unsigned short* __restrict__ A, int lda,
                                            const unsigned short* __restrict__ B0,
                                            const unsigned short* __restrict__ B1,
                                            const unsigned short* __restrict__ B2,
                                            const unsigned short* __restrict__ B3,
                                            int ldb, int lane, int k0) {
    FragSet s;
    s.a  = load_frag16x32(A,  lda, lane, k0);
    s.b0 = load_frag16x32(B0, ldb, lane, k0);
    s.b1 = load_frag16x32(B1, ldb, lane, k0);
    s.b2 = load_frag16x32(B2, ldb, lane, k0);
    s.b3 = load_frag16x32(B3, ldb, lane, k0);
    return s;
}

__device__ __forceinline__ void mma_set(const FragSet& s, v8f acc[4]) {
    acc[0] = __builtin_amdgcn_wmma_f32_16x16x32_bf16(false, s.a, false, s.b0, (short)0, acc[0], false, false);
    acc[1] = __builtin_amdgcn_wmma_f32_16x16x32_bf16(false, s.a, false, s.b1, (short)0, acc[1], false, false);
    acc[2] = __builtin_amdgcn_wmma_f32_16x16x32_bf16(false, s.a, false, s.b2, (short)0, acc[2], false, false);
    acc[3] = __builtin_amdgcn_wmma_f32_16x16x32_bf16(false, s.a, false, s.b3, (short)0, acc[3], false, false);
}

// Pipelined GEMM over one contiguous K-segment (len % 64 == 0).
// Rotated + 2x-unrolled: while the 4 WMMAs of one chunk execute, the next
// chunk's 10 B128 loads are in flight. Tail chunk peeled -> branch-free body.
__device__ __forceinline__ void seg_gemm(const unsigned short* __restrict__ A, int lda,
                                         const unsigned short* __restrict__ B0,
                                         const unsigned short* __restrict__ B1,
                                         const unsigned short* __restrict__ B2,
                                         const unsigned short* __restrict__ B3,
                                         int ldb, int len, int lane, v8f acc[4]) {
    FragSet s0 = load_set(A, lda, B0, B1, B2, B3, ldb, lane, 0);
    int k0 = 0;
    for (; k0 + 64 < len; k0 += 64) {
        FragSet s1 = load_set(A, lda, B0, B1, B2, B3, ldb, lane, k0 + 32);
        mma_set(s0, acc);
        s0 = load_set(A, lda, B0, B1, B2, B3, ldb, lane, k0 + 64);
        mma_set(s1, acc);
    }
    FragSet s1 = load_set(A, lda, B0, B1, B2, B3, ldb, lane, k0 + 32);
    mma_set(s0, acc);
    mma_set(s1, acc);
}

// ---------------------------------------------------------------------------
// Prep kernels
// ---------------------------------------------------------------------------
__global__ void convert_f32_bf16(const float* __restrict__ src,
                                 unsigned short* __restrict__ dst, int n) {
    int i = blockIdx.x * blockDim.x + threadIdx.x;
    if (i < n) dst[i] = f32_to_bf16(src[i]);
}

// src: [K x N] fp32 row-major -> dst: [N x K] bf16 row-major (transpose).
// Block (32,8), grid (N/32, K/32). LDS-tiled for coalescing both sides.
__global__ void transpose_convert(const float* __restrict__ src,
                                  unsigned short* __restrict__ dst,
                                  int K, int N) {
    __shared__ float tile[32][33];
    int n0 = blockIdx.x * 32, k0 = blockIdx.y * 32;
    int tx = threadIdx.x, ty = threadIdx.y;
    for (int i = ty; i < 32; i += 8)
        tile[i][tx] = src[(size_t)(k0 + i) * N + n0 + tx];
    __syncthreads();
    for (int i = ty; i < 32; i += 8)
        dst[(size_t)(n0 + i) * K + k0 + tx] = f32_to_bf16(tile[tx][i]);
}

// ---------------------------------------------------------------------------
// Fused LSTM layer step:  G = [xa|xb] @ W + bias; gate math; c,h update.
// block = 128 (4 waves); wave w owns hidden tile blockIdx.x*4+w.
// grid = (HID/64, BATCH/16).
// ---------------------------------------------------------------------------
__global__ __launch_bounds__(128) void lstm_step(
    const unsigned short* __restrict__ xa, int Ka, int lda,   // input part 1 (bf16)
    const unsigned short* __restrict__ xb, int Kb, int ldb,   // input part 2 = h_prev
    const unsigned short* __restrict__ Wt,                    // [4H x K] bf16 (pre-transposed)
    const float* __restrict__ bias,                           // [4H]
    float* __restrict__ c,                                    // [B x H] in/out
    unsigned short* __restrict__ hout,                        // [B x H] bf16
    unsigned short* __restrict__ hout2)                       // optional copy (outs row), or null
{
    const int lane  = threadIdx.x & 31;
    const int ntile = blockIdx.x * 4 + (threadIdx.x >> 5);    // hidden tile
    const int mtile = blockIdx.y;                             // batch tile
    const int K = Ka + Kb;

    v8f acc[4] = {};
    const unsigned short* Aa = xa + (size_t)mtile * 16 * lda;
    const unsigned short* Ab = xb + (size_t)mtile * 16 * ldb;
    const unsigned short* B0 = Wt + (size_t)(0 * HID + ntile * 16) * K;
    const unsigned short* B1 = Wt + (size_t)(1 * HID + ntile * 16) * K;
    const unsigned short* B2 = Wt + (size_t)(2 * HID + ntile * 16) * K;
    const unsigned short* B3 = Wt + (size_t)(3 * HID + ntile * 16) * K;

    seg_gemm(Aa, lda, B0, B1, B2, B3, K, Ka, lane, acc);                     // [x | .]
    seg_gemm(Ab, ldb, B0 + Ka, B1 + Ka, B2 + Ka, B3 + Ka, K, Kb, lane, acc); // [. | h]

    // Epilogue: C/D layout -> lane l, VGPR r holds (m = r + 8*(l>>4), n = l&15)
    const int n   = lane & 15;
    const int sel = lane >> 4;
    const int col = ntile * 16 + n;
    const float bi = bias[0 * HID + col];
    const float bf = bias[1 * HID + col];
    const float bg = bias[2 * HID + col];
    const float bo = bias[3 * HID + col];
#pragma unroll
    for (int r = 0; r < 8; ++r) {
        int b = mtile * 16 + r + 8 * sel;
        size_t idx = (size_t)b * HID + col;
        float vi = sigmoidf_(acc[0][r] + bi);
        float vf = sigmoidf_(acc[1][r] + bf);
        float vg = tanhf_  (acc[2][r] + bg);
        float vo = sigmoidf_(acc[3][r] + bo);
        float cn = vf * c[idx] + vi * vg;
        c[idx] = cn;
        unsigned short hb = f32_to_bf16(vo * tanhf_(cn));
        hout[idx] = hb;
        if (hout2) hout2[idx] = hb;
    }
}

// ---------------------------------------------------------------------------
// Final projection: out[b,t,d] = sigmoid(outs[t*B+b,:] . finWt[d,:] + fb[d])
// block = 128 (4 waves); grid = (DIN/64, FUT*BATCH/16).
// Same rotated 2x-unrolled pipeline with A/B pair sets.
// ---------------------------------------------------------------------------
__global__ __launch_bounds__(128) void final_proj(
    const unsigned short* __restrict__ outs,   // [(FUT*B) x H] bf16
    const unsigned short* __restrict__ Wt,     // [D x H] bf16 (pre-transposed)
    const float* __restrict__ bias,            // [D]
    float* __restrict__ out)                   // [B, FUT, D] fp32
{
    const int lane  = threadIdx.x & 31;
    const int ntile = blockIdx.x * 4 + (threadIdx.x >> 5);
    const int mtile = blockIdx.y;

    v8f acc = {};
    const unsigned short* Arow = outs + (size_t)mtile * 16 * HID;
    const unsigned short* Brow = Wt   + (size_t)ntile * 16 * HID;

    v16bf a0 = load_frag16x32(Arow, HID, lane, 0);
    v16bf b0 = load_frag16x32(Brow, HID, lane, 0);
    int k0 = 0;
    for (; k0 + 64 < HID; k0 += 64) {
        v16bf a1 = load_frag16x32(Arow, HID, lane, k0 + 32);
        v16bf b1 = load_frag16x32(Brow, HID, lane, k0 + 32);
        acc = __builtin_amdgcn_wmma_f32_16x16x32_bf16(false, a0, false, b0, (short)0, acc, false, false);
        a0 = load_frag16x32(Arow, HID, lane, k0 + 64);
        b0 = load_frag16x32(Brow, HID, lane, k0 + 64);
        acc = __builtin_amdgcn_wmma_f32_16x16x32_bf16(false, a1, false, b1, (short)0, acc, false, false);
    }
    {
        v16bf a1 = load_frag16x32(Arow, HID, lane, k0 + 32);
        v16bf b1 = load_frag16x32(Brow, HID, lane, k0 + 32);
        acc = __builtin_amdgcn_wmma_f32_16x16x32_bf16(false, a0, false, b0, (short)0, acc, false, false);
        acc = __builtin_amdgcn_wmma_f32_16x16x32_bf16(false, a1, false, b1, (short)0, acc, false, false);
    }
    const int n   = lane & 15;
    const int sel = lane >> 4;
    const int d   = ntile * 16 + n;
    const float bb = bias[d];
#pragma unroll
    for (int r = 0; r < 8; ++r) {
        int row = mtile * 16 + r + 8 * sel;   // row = t*B + b
        int t = row >> 6;                     // / BATCH
        int b = row & 63;
        out[((size_t)b * FUT + t) * DIN + d] = sigmoidf_(acc[r] + bb);
    }
}

// ---------------------------------------------------------------------------
// Host orchestration (graph-capture safe: only async ops on `stream`).
// Workspace requirement: ~92 MB.
// ---------------------------------------------------------------------------
extern "C" void kernel_launch(void* const* d_in, const int* in_sizes, int n_in,
                              void* d_out, int out_size, void* d_ws, size_t ws_size,
                              hipStream_t stream) {
    const float* x      = (const float*)d_in[0];
    const float* encW0  = (const float*)d_in[1];
    const float* encB0  = (const float*)d_in[2];
    const float* encW1  = (const float*)d_in[3];
    const float* encB1  = (const float*)d_in[4];
    const float* decW0  = (const float*)d_in[5];
    const float* decB0  = (const float*)d_in[6];
    const float* decW1  = (const float*)d_in[7];
    const float* decB1  = (const float*)d_in[8];
    const float* finW   = (const float*)d_in[9];
    const float* finB   = (const float*)d_in[10];
    float* out = (float*)d_out;

    // ---- workspace carve-up (256B aligned) ----
    size_t off = 0;
    auto carve = [&](size_t bytes) -> void* {
        off = (off + 255) & ~(size_t)255;
        void* p = (char*)d_ws + off;
        off += bytes;
        return p;
    };
    unsigned short* xbf    = (unsigned short*)carve((size_t)BATCH * SEQT * DIN * 2);
    unsigned short* e0Wt   = (unsigned short*)carve((size_t)4 * HID * (DIN + HID) * 2);
    unsigned short* e1Wt   = (unsigned short*)carve((size_t)4 * HID * (2 * HID) * 2);
    unsigned short* d0Wt   = (unsigned short*)carve((size_t)4 * HID * (2 * HID) * 2);
    unsigned short* d1Wt   = (unsigned short*)carve((size_t)4 * HID * (2 * HID) * 2);
    unsigned short* fWt    = (unsigned short*)carve((size_t)DIN * HID * 2);
    unsigned short* outs   = (unsigned short*)carve((size_t)FUT * BATCH * HID * 2);
    unsigned short* h0e[2], *h1e[2], *h0d[2], *h1d[2];
    for (int i = 0; i < 2; ++i) h0e[i] = (unsigned short*)carve((size_t)BATCH * HID * 2);
    for (int i = 0; i < 2; ++i) h1e[i] = (unsigned short*)carve((size_t)BATCH * HID * 2);
    for (int i = 0; i < 2; ++i) h0d[i] = (unsigned short*)carve((size_t)BATCH * HID * 2);
    for (int i = 0; i < 2; ++i) h1d[i] = (unsigned short*)carve((size_t)BATCH * HID * 2);
    float* c0e = (float*)carve((size_t)BATCH * HID * 4);
    float* c1e = (float*)carve((size_t)BATCH * HID * 4);
    float* c0d = (float*)carve((size_t)BATCH * HID * 4);
    float* c1d = (float*)carve((size_t)BATCH * HID * 4);
    (void)ws_size; (void)n_in; (void)in_sizes; (void)out_size;

    // ---- zero recurrent state (deterministic per call) ----
    for (int i = 0; i < 2; ++i) {
        hipMemsetAsync(h0e[i], 0, (size_t)BATCH * HID * 2, stream);
        hipMemsetAsync(h1e[i], 0, (size_t)BATCH * HID * 2, stream);
        hipMemsetAsync(h0d[i], 0, (size_t)BATCH * HID * 2, stream);
        hipMemsetAsync(h1d[i], 0, (size_t)BATCH * HID * 2, stream);
    }
    hipMemsetAsync(c0e, 0, (size_t)BATCH * HID * 4, stream);
    hipMemsetAsync(c1e, 0, (size_t)BATCH * HID * 4, stream);
    hipMemsetAsync(c0d, 0, (size_t)BATCH * HID * 4, stream);
    hipMemsetAsync(c1d, 0, (size_t)BATCH * HID * 4, stream);

    // ---- one-time conversions: x -> bf16, weights -> bf16 [N x K] ----
    {
        int n = BATCH * SEQT * DIN;
        convert_f32_bf16<<<(n + 255) / 256, 256, 0, stream>>>(x, xbf, n);
    }
    dim3 tb(32, 8);
    transpose_convert<<<dim3(4 * HID / 32, (DIN + HID) / 32), tb, 0, stream>>>(encW0, e0Wt, DIN + HID, 4 * HID);
    transpose_convert<<<dim3(4 * HID / 32, (2 * HID) / 32), tb, 0, stream>>>(encW1, e1Wt, 2 * HID, 4 * HID);
    transpose_convert<<<dim3(4 * HID / 32, (2 * HID) / 32), tb, 0, stream>>>(decW0, d0Wt, 2 * HID, 4 * HID);
    transpose_convert<<<dim3(4 * HID / 32, (2 * HID) / 32), tb, 0, stream>>>(decW1, d1Wt, 2 * HID, 4 * HID);
    transpose_convert<<<dim3(DIN / 32, HID / 32), tb, 0, stream>>>(finW, fWt, HID, DIN);

    const dim3 sgrid(HID / 64, BATCH / 16);   // 16 x 4 blocks, 4 waves each
    const dim3 sblk(128);

    // ---- encoder: 128 steps x 2 stacked cells (stream-ordered dependency) ----
    for (int t = 0; t < SEQT; ++t) {
        int pr = t & 1, pw = pr ^ 1;
        // layer 0: input = [x_t | h0_prev]
        lstm_step<<<sgrid, sblk, 0, stream>>>(
            xbf + (size_t)t * DIN, DIN, SEQT * DIN,
            h0e[pr], HID, HID,
            e0Wt, encB0, c0e, h0e[pw], (unsigned short*)nullptr);
        // layer 1: input = [h0_new | h1_prev]
        lstm_step<<<sgrid, sblk, 0, stream>>>(
            h0e[pw], HID, HID,
            h1e[pr], HID, HID,
            e1Wt, encB1, c1e, h1e[pw], (unsigned short*)nullptr);
    }
    unsigned short* enc_state = h1e[SEQT & 1];   // final write parity: (T-1)&1 ^ 1

    // ---- decoder: 128 steps, feeds its own top-layer output back ----
    for (int t = 0; t < FUT; ++t) {
        int pr = t & 1, pw = pr ^ 1;
        const unsigned short* s = (t == 0) ? enc_state : h1d[pr];
        lstm_step<<<sgrid, sblk, 0, stream>>>(
            s, HID, HID,
            h0d[pr], HID, HID,
            d0Wt, decB0, c0d, h0d[pw], (unsigned short*)nullptr);
        lstm_step<<<sgrid, sblk, 0, stream>>>(
            h0d[pw], HID, HID,
            h1d[pr], HID, HID,
            d1Wt, decB1, c1d, h1d[pw],
            outs + (size_t)t * BATCH * HID);      // also record outs[t] row-block
    }

    // ---- final projection + sigmoid + (t,b)->(b,t) transpose ----
    final_proj<<<dim3(DIN / 64, FUT * BATCH / 16), sblk, 0, stream>>>(outs, fWt, finB, out);
}
